// GAT_30279519437045
// MI455X (gfx1250) — matrix operand order repaired and checked by reading, not compiled, not run on hardware
//
#include <hip/hip_runtime.h>

// ---------------- problem constants (match reference setup) ----------------
#define NNODES   4096
#define KEDGE    32
#define NFEAT    128
#define NHEADS   4
#define NHID     128
#define NHD      512      // NHEADS * NHID
#define NCLASS   32
#define NODE_NUM 1024
#define ALPHA    0.2f

// ---------------- workspace layout (bytes, 256-aligned) ----------------
#define OFF_MASK  ((size_t)0)                                   // [4096][32] u32   = 512KB
#define OFF_XBF   (OFF_MASK + (size_t)NNODES*32*4)              // [4096][128] bf16 = 1MB
#define OFF_WSBP  (OFF_XBF  + (size_t)NNODES*NFEAT*2)           // packed Ws frags  = 128KB
#define OFF_WOBP  (OFF_WSBP + (size_t)NHEADS*NFEAT*NHID*2)      // packed W_out frags = 32KB
#define OFF_H     (OFF_WOBP + (size_t)NHD*NCLASS*2)             // [4][4096][128] f32 = 8MB
#define OFF_F1    (OFF_H    + (size_t)NHEADS*NNODES*NHID*4)     // [4][4096] f32
#define OFF_F2    (OFF_F1   + (size_t)NHEADS*NNODES*4)          // [4][4096] f32
#define OFF_XCBF  (OFF_F2   + (size_t)NHEADS*NNODES*4)          // [4096][512] bf16 = 4MB
#define OFF_H2    (OFF_XCBF + (size_t)NNODES*NHD*2)             // [4096][32] f32
#define OFF_F1O   (OFF_H2   + (size_t)NNODES*NCLASS*4)          // [4096] f32
#define OFF_F2O   (OFF_F1O  + (size_t)NNODES*4)                 // [4096] f32

// ---------------- types ----------------
typedef __attribute__((ext_vector_type(16))) __bf16 v16bf;
typedef __attribute__((ext_vector_type(8)))  float  v8f;

struct alignas(16) BF16Frag { unsigned short u[16]; };

// ---------------- helpers ----------------
__device__ __forceinline__ unsigned short f2bf(float f) {
    unsigned u = __builtin_bit_cast(unsigned, f);
    u += 0x7fffu + ((u >> 16) & 1u);          // round-to-nearest-even
    return (unsigned short)(u >> 16);
}
__device__ __forceinline__ float lrelu(float x) { return x > 0.f ? x : ALPHA * x; }
__device__ __forceinline__ float eluf (float x) { return x > 0.f ? x : __expf(x) - 1.f; }

// A fragment: 16x32 bf16, M x K.  lanes 0-15: row m0+lane, K = k0+[0..7],[16..23]
//                                 lanes 16-31: row m0+lane-16, K = k0+[8..15],[24..31]
__device__ __forceinline__ v16bf load_a_frag(const unsigned short* A, int lda,
                                             int m0, int k0, int lane) {
    const unsigned short* p = A + (size_t)(m0 + (lane & 15)) * lda + k0 + ((lane & 16) ? 8 : 0);
    BF16Frag f;
    *(uint4*)(f.u)     = *(const uint4*)(p);        // contiguous 8 bf16
    *(uint4*)(f.u + 8) = *(const uint4*)(p + 16);   // contiguous 8 bf16
    return __builtin_bit_cast(v16bf, f);
}

// Pre-packed B fragment: 32 contiguous bytes per lane (two b128 loads).
__device__ __forceinline__ v16bf load_bp_frag(const unsigned short* BP, int fragIdx, int lane) {
    const unsigned short* p = BP + ((size_t)fragIdx * 32 + lane) * 16;
    BF16Frag f;
    *(uint4*)(f.u)     = *(const uint4*)(p);
    *(uint4*)(f.u + 8) = *(const uint4*)(p + 8);
    return __builtin_bit_cast(v16bf, f);
}

// C/D 16x16 f32: VGPR r -> M = r (lanes 0-15) / r+8 (lanes 16-31); N = lane&15.
__device__ __forceinline__ void store_c(float* C, int ldc, int m0, int n0, int lane, v8f acc) {
    int n  = n0 + (lane & 15);
    int mb = m0 + ((lane & 16) ? 8 : 0);
#pragma unroll
    for (int r = 0; r < 8; ++r) C[(size_t)(mb + r) * ldc + n] = acc[r];
}

// ---------------- kernels ----------------
__global__ void k_zero_u32(unsigned* p, int n) {
    int t = blockIdx.x * blockDim.x + threadIdx.x;
    if (t < n) p[t] = 0u;
}

__global__ void k_build_mask(const int* __restrict__ edge, unsigned* __restrict__ mask) {
    int t = blockIdx.x * blockDim.x + threadIdx.x;      // over N*KEDGE
    if (t >= NNODES * KEDGE) return;
    int e = edge[t];
    if (e >= 0 && e < NODE_NUM) {
        int r = t >> 5;                                 // KEDGE == 32
        atomicOr(&mask[r * 32 + (e >> 5)], 1u << (e & 31));
    }
}

__global__ void k_cvt_bf16(const float* __restrict__ in, unsigned short* __restrict__ out, int n) {
    int t = blockIdx.x * blockDim.x + threadIdx.x;
    if (t < n) out[t] = f2bf(in[t]);
}

// Convert + pack a row-major f32 B matrix [K][Ncols] into WMMA B-fragment order:
// BP[((kc*NT + nt)*32 + lane)*16 + t] = B[kc*32 + (lane&16?8:0) + (t&7) + (t&8?16:0)][nt*16 + (lane&15)]
__global__ void k_pack_b(const float* __restrict__ B, unsigned short* __restrict__ BP,
                         int Ncols, int NT, int total) {
    int idx = blockIdx.x * blockDim.x + threadIdx.x;
    if (idx >= total) return;
    int t    = idx & 15;
    int lane = (idx >> 4) & 31;
    int frag = idx >> 9;
    int nt   = frag % NT;
    int kc   = frag / NT;
    int n    = nt * 16 + (lane & 15);
    int k    = kc * 32 + ((lane & 16) ? 8 : 0) + (t & 7) + ((t & 8) ? 16 : 0);
    BP[idx] = f2bf(B[(size_t)k * Ncols + n]);
}

// h[head] = x @ Ws[head] : [4096,128]x[128,128] bf16 WMMA, f32 accum.
// 4 waves/block; each wave computes two adjacent 16-wide N-tiles (A frag reused).
__global__ void k_gemm1(const unsigned short* __restrict__ xbf,
                        const unsigned short* __restrict__ wsbp,
                        float* __restrict__ h) {
    int lane = threadIdx.x & 31;
    int wave = threadIdx.x >> 5;          // 4 waves: N-tile pairs
    int m0   = blockIdx.x * 16;
    int head = blockIdx.y;
    const unsigned short* BP = wsbp + (size_t)head * NFEAT * NHID;  // 32 frags/head
    v8f acc0 = {}, acc1 = {};
#pragma unroll
    for (int kc = 0; kc < NFEAT / 32; ++kc) {
        v16bf a  = load_a_frag(xbf, NFEAT, m0, kc * 32, lane);
        v16bf b0 = load_bp_frag(BP, kc * 8 + wave * 2 + 0, lane);
        v16bf b1 = load_bp_frag(BP, kc * 8 + wave * 2 + 1, lane);
        acc0 = __builtin_amdgcn_wmma_f32_16x16x32_bf16(false, a, false, b0,
                                                       (short)0, acc0, false, false);
        acc1 = __builtin_amdgcn_wmma_f32_16x16x32_bf16(false, a, false, b1,
                                                       (short)0, acc1, false, false);
    }
    float* Ch = h + (size_t)head * NNODES * NHID;
    store_c(Ch, NHID, m0, wave * 32,      lane, acc0);
    store_c(Ch, NHID, m0, wave * 32 + 16, lane, acc1);
}

// f1/f2 = H @ a[:F], H @ a[F:]  (generic over layers; heads laid out contiguously)
__global__ void k_fvec(const float* __restrict__ Hm, const float* __restrict__ avec,
                       float* __restrict__ f1, float* __restrict__ f2,
                       int Fdim, int rows_per_head, int total) {
    int t = blockIdx.x * blockDim.x + threadIdx.x;
    if (t >= total) return;
    int head = t / rows_per_head;
    const float* hr = Hm + (size_t)t * Fdim;
    const float* a1 = avec + (size_t)head * 2 * Fdim;
    const float* a2 = a1 + Fdim;
    float s1 = 0.f, s2 = 0.f;
    for (int j = 0; j < Fdim; ++j) { float v = hr[j]; s1 += v * a1[j]; s2 += v * a2[j]; }
    f1[t] = s1; f2[t] = s2;
}

// Sparse masked softmax attention, layer 1. One wave per (head,row); lanes split 128 feats.
// Writes xc directly in concat layout [4096][512] as bf16.
__global__ void k_attn1(const unsigned* __restrict__ mask,
                        const float* __restrict__ f1, const float* __restrict__ f2,
                        const float* __restrict__ h, unsigned short* __restrict__ xcbf) {
    int wid  = (blockIdx.x * blockDim.x + threadIdx.x) >> 5;
    int lane = threadIdx.x & 31;
    int head = wid >> 12;                 // 4096 rows per head
    int r    = wid & (NNODES - 1);
    int base = (r >> 10) << 10;           // block start
    const unsigned* mrow = mask + (size_t)r * 32;
    const float* f2h = f2 + (size_t)head * NNODES;
    float f1v = f1[(size_t)head * NNODES + r];

    // pass 1: row max of e = lrelu(f1+f2[j]) (monotonic -> use max f2)
    float mx = -3.0e38f;
    for (int w = 0; w < 32; ++w) {
        unsigned bits = mrow[w];
        while (bits) {
            int b = __builtin_ctz(bits); bits &= bits - 1;
            mx = fmaxf(mx, f2h[base + (w << 5) + b]);
        }
    }
    float me = lrelu(f1v + mx);

    // pass 2: accumulate softmax-weighted neighbor features (float4 per lane)
    float den = 0.f;
    float ax = 0.f, ay = 0.f, az = 0.f, aw = 0.f;
    const float4* hb = (const float4*)(h + (size_t)head * NNODES * NHID);
    for (int w = 0; w < 32; ++w) {
        unsigned bits = mrow[w];
        while (bits) {
            int b = __builtin_ctz(bits); bits &= bits - 1;
            int j = base + (w << 5) + b;
            float p = __expf(lrelu(f1v + f2h[j]) - me);
            den += p;
            float4 v = hb[(size_t)j * 32 + lane];
            ax += p * v.x; ay += p * v.y; az += p * v.z; aw += p * v.w;
        }
    }
    float inv = 1.0f / den;
    float o0 = eluf(ax * inv), o1 = eluf(ay * inv), o2 = eluf(az * inv), o3 = eluf(aw * inv);
    unsigned p0 = (unsigned)f2bf(o0) | ((unsigned)f2bf(o1) << 16);
    unsigned p1 = (unsigned)f2bf(o2) | ((unsigned)f2bf(o3) << 16);
    unsigned short* op = xcbf + (size_t)r * NHD + head * NHID + lane * 4;
    *(uint2*)op = make_uint2(p0, p1);
}

// h2 = xc @ W_out : [4096,512]x[512,32] via bf16 WMMA
__global__ void k_gemm2(const unsigned short* __restrict__ xcbf,
                        const unsigned short* __restrict__ wobp,
                        float* __restrict__ h2) {
    int lane = threadIdx.x & 31;
    int wave = threadIdx.x >> 5;          // 2 waves: N-tiles 0..1
    int m0   = blockIdx.x * 16;
    v8f acc = {};
#pragma unroll
    for (int kc = 0; kc < NHD / 32; ++kc) {
        v16bf a = load_a_frag(xcbf, NHD, m0, kc * 32, lane);
        v16bf b = load_bp_frag(wobp, kc * 2 + wave, lane);
        acc = __builtin_amdgcn_wmma_f32_16x16x32_bf16(false, a, false, b,
                                                      (short)0, acc, false, false);
    }
    store_c(h2, NCLASS, m0, wave * 16, lane, acc);
}

// Sparse attention layer 2 + elu + log_softmax. One wave per row, lane = class.
__global__ void k_attn2(const unsigned* __restrict__ mask,
                        const float* __restrict__ f1o, const float* __restrict__ f2o,
                        const float* __restrict__ h2, float* __restrict__ out) {
    int wid  = (blockIdx.x * blockDim.x + threadIdx.x) >> 5;
    int lane = threadIdx.x & 31;
    int r    = wid;
    int base = (r >> 10) << 10;
    const unsigned* mrow = mask + (size_t)r * 32;
    float f1v = f1o[r];

    float mx = -3.0e38f;
    for (int w = 0; w < 32; ++w) {
        unsigned bits = mrow[w];
        while (bits) {
            int b = __builtin_ctz(bits); bits &= bits - 1;
            mx = fmaxf(mx, f2o[base + (w << 5) + b]);
        }
    }
    float me = lrelu(f1v + mx);

    float den = 0.f, acc = 0.f;
    for (int w = 0; w < 32; ++w) {
        unsigned bits = mrow[w];
        while (bits) {
            int b = __builtin_ctz(bits); bits &= bits - 1;
            int j = base + (w << 5) + b;
            float p = __expf(lrelu(f1v + f2o[j]) - me);
            den += p;
            acc += p * h2[(size_t)j * NCLASS + lane];
        }
    }
    float v = eluf(acc / den);

    // log_softmax across the 32 lanes (wave32)
    float m2 = v;
#pragma unroll
    for (int off = 16; off; off >>= 1) m2 = fmaxf(m2, __shfl_xor(m2, off, 32));
    float s = __expf(v - m2);
#pragma unroll
    for (int off = 16; off; off >>= 1) s += __shfl_xor(s, off, 32);
    out[(size_t)r * NCLASS + lane] = v - m2 - __logf(s);
}

// ---------------- launch ----------------
extern "C" void kernel_launch(void* const* d_in, const int* in_sizes, int n_in,
                              void* d_out, int out_size, void* d_ws, size_t ws_size,
                              hipStream_t stream) {
    (void)in_sizes; (void)n_in; (void)out_size; (void)ws_size;
    const float* x     = (const float*)d_in[0];
    const int*   edge  = (const int*)d_in[1];
    // d_in[2] = node_num (=1024, compiled in)
    const float* Ws    = (const float*)d_in[3];
    const float* avec  = (const float*)d_in[4];
    const float* W_out = (const float*)d_in[5];
    const float* a_out = (const float*)d_in[6];
    float* out = (float*)d_out;

    char* ws = (char*)d_ws;
    unsigned*       mask  = (unsigned*)      (ws + OFF_MASK);
    unsigned short* xbf   = (unsigned short*)(ws + OFF_XBF);
    unsigned short* wsbp  = (unsigned short*)(ws + OFF_WSBP);
    unsigned short* wobp  = (unsigned short*)(ws + OFF_WOBP);
    float*          h     = (float*)         (ws + OFF_H);
    float*          f1    = (float*)         (ws + OFF_F1);
    float*          f2    = (float*)         (ws + OFF_F2);
    unsigned short* xcbf  = (unsigned short*)(ws + OFF_XCBF);
    float*          h2    = (float*)         (ws + OFF_H2);
    float*          f1o   = (float*)         (ws + OFF_F1O);
    float*          f2o   = (float*)         (ws + OFF_F2O);

    // 1) adjacency bitmask (dedups duplicate edges, matches dense adj>0)
    k_zero_u32 <<<dim3(512), dim3(256), 0, stream>>>(mask, NNODES * 32);
    k_build_mask<<<dim3(512), dim3(256), 0, stream>>>(edge, mask);

    // 2) bf16 conversion of x; weights converted + packed into B-fragment order
    k_cvt_bf16<<<dim3(2048), dim3(256), 0, stream>>>(x, xbf, NNODES * NFEAT);
    for (int hd = 0; hd < NHEADS; ++hd) {
        k_pack_b<<<dim3(64), dim3(256), 0, stream>>>(Ws + (size_t)hd * NFEAT * NHID,
                                                     wsbp + (size_t)hd * NFEAT * NHID,
                                                     NHID, NHID / 16, NFEAT * NHID);
    }
    k_pack_b<<<dim3(64), dim3(256), 0, stream>>>(W_out, wobp, NCLASS, NCLASS / 16, NHD * NCLASS);

    // 3) per-head projection h = x @ Ws[head]  (WMMA bf16 -> f32)
    k_gemm1<<<dim3(NNODES / 16, NHEADS), dim3(128), 0, stream>>>(xbf, wsbp, h);

    // 4) attention logits halves f1/f2
    k_fvec<<<dim3(64), dim3(256), 0, stream>>>(h, avec, f1, f2, NHID, NNODES, NHEADS * NNODES);

    // 5) sparse masked softmax attention + elu -> xc (bf16, concat layout)
    k_attn1<<<dim3(2048), dim3(256), 0, stream>>>(mask, f1, f2, h, xcbf);

    // 6) output projection h2 = xc @ W_out  (WMMA bf16 -> f32)
    k_gemm2<<<dim3(NNODES / 16), dim3(64), 0, stream>>>(xcbf, wobp, h2);

    // 7) output-layer logits halves
    k_fvec<<<dim3(16), dim3(256), 0, stream>>>(h2, a_out, f1o, f2o, NCLASS, NNODES, NNODES);

    // 8) sparse attention + elu + log_softmax -> out
    k_attn2<<<dim3(512), dim3(256), 0, stream>>>(mask, f1o, f2o, h2, out);
}
